// NodeRemovalNet_16544214024641
// MI455X (gfx1250) — compile-verified, matrix operand order checked by reading, not run on hardware
//
#include <hip/hip_runtime.h>
#include <hip/hip_bf16.h>

typedef __attribute__((ext_vector_type(16))) _Float16 v16h;
typedef __attribute__((ext_vector_type(8)))  float    v8f;
typedef _Float16 h16;
typedef unsigned long long u64;

constexpr int B_   = 8;
constexpr int NG_  = 16384;
constexpr int DEG_ = 16;
constexpr int N0_  = B_ * NG_;        // 131072
constexpr int E_   = B_ * NG_ * DEG_; // 2097152
constexpr int N1_  = N0_ / 2;         // 65536
constexpr int N2_  = N0_ / 4;         // 32768
constexpr int N4_  = N0_ / 8;         // 16384

// ---------------------------------------------------------------------------
// Weight f32 -> f16 conversion (once per launch, 64x64 each)
// ---------------------------------------------------------------------------
__global__ void cvt_w64(const float* __restrict__ w, h16* __restrict__ o) {
  int i = blockIdx.x * blockDim.x + threadIdx.x;   // 4096 elements
  o[i] = (h16)w[i];
}

// ---------------------------------------------------------------------------
// SAGE layer 1: in-dim 2 (scalar path, K too small for WMMA)
// ---------------------------------------------------------------------------
__global__ void sage1_edge(const float* __restrict__ x, const int* __restrict__ src,
                           const int* __restrict__ dst, float* __restrict__ agg2,
                           float* __restrict__ cnt) {
  int e = blockIdx.x * blockDim.x + threadIdx.x;
  if (e >= E_) return;
  int s = src[e], d = dst[e];
  atomicAdd(&agg2[d * 2 + 0], x[s * 2 + 0]);
  atomicAdd(&agg2[d * 2 + 1], x[s * 2 + 1]);
  atomicAdd(&cnt[d], 1.0f);
}

__global__ void sage1_node(const float* __restrict__ x, const float* __restrict__ agg2,
                           const float* __restrict__ cnt, const float* __restrict__ Wl,
                           const float* __restrict__ bl, const float* __restrict__ Wr,
                           h16* __restrict__ out) {
  int n = blockIdx.x * blockDim.x + threadIdx.x;
  if (n >= N0_) return;
  float c  = fmaxf(cnt[n], 1.0f);
  float m0 = agg2[n * 2 + 0] / c, m1 = agg2[n * 2 + 1] / c;
  float x0 = x[n * 2 + 0], x1 = x[n * 2 + 1];
  #pragma unroll 8
  for (int j = 0; j < 64; ++j) {
    float v = m0 * Wl[j * 2] + m1 * Wl[j * 2 + 1] + bl[j] +
              x0 * Wr[j * 2] + x1 * Wr[j * 2 + 1];
    out[n * 64 + j] = (h16)fmaxf(v, 0.0f);
  }
}

// ---------------------------------------------------------------------------
// TopK pooling: score+key -> rank (one u64 compare per candidate) -> compact
// key = (sortable_float(score) << 32) | (n-1-localIdx)  => descending order
// with ties broken by lower index, exactly matching jax.lax.top_k.
// ---------------------------------------------------------------------------
__global__ void score_kernel(const h16* __restrict__ hft, const float* __restrict__ w,
                             float* __restrict__ score, u64* __restrict__ keys, int n) {
  int b  = blockIdx.y;
  int il = blockIdx.x * blockDim.x + threadIdx.x;
  int gi = b * n + il;
  float dot = 0.0f, nrm = 0.0f;
  #pragma unroll 8
  for (int f = 0; f < 64; ++f) {
    dot += (float)hft[gi * 64 + f] * w[f];
    nrm += w[f] * w[f];
  }
  float s = tanhf(dot * rsqrtf(nrm));
  score[gi] = s;
  unsigned u = __float_as_uint(s);
  u = (u & 0x80000000u) ? ~u : (u | 0x80000000u);    // sortable ascending
  keys[gi] = ((u64)u << 32) | (unsigned)(n - 1 - il);
}

__global__ void rank_kernel(const u64* __restrict__ keys, int* __restrict__ rank, int n) {
  __shared__ u64 sh[1024];
  int b  = blockIdx.y;
  int il = blockIdx.x * blockDim.x + threadIdx.x;
  u64 ki = keys[b * n + il];
  int r = 0;
  for (int base = 0; base < n; base += 1024) {
    __syncthreads();
    for (int t = threadIdx.x; t < 1024; t += blockDim.x)
      sh[t] = keys[b * n + base + t];
    __syncthreads();
    #pragma unroll 8
    for (int j = 0; j < 1024; ++j) r += (sh[j] > ki);
  }
  rank[b * n + il] = r;
}

__global__ void topk_compact(const h16* __restrict__ hin, const float* __restrict__ score,
                             const int* __restrict__ rank, h16* __restrict__ hout,
                             int* __restrict__ npos, int n, int k) {
  int b = blockIdx.y;
  int i = blockIdx.x * blockDim.x + threadIdx.x;
  int gi = b * n + i;
  int r = rank[gi];
  if (r < k) {
    int ni = b * k + r;
    npos[gi] = ni;
    float sv = score[gi];
    #pragma unroll 8
    for (int f = 0; f < 64; ++f)
      hout[ni * 64 + f] = (h16)((float)hin[gi * 64 + f] * sv);
  } else {
    npos[gi] = -1;
  }
}

template <bool FIRST>
__global__ void edge_remap(const int* __restrict__ src, const int* __restrict__ dst,
                           const float* __restrict__ mask, const int* __restrict__ npos,
                           int* __restrict__ nsrc, int* __restrict__ ndst,
                           float* __restrict__ nmask) {
  int e = blockIdx.x * blockDim.x + threadIdx.x;
  if (e >= E_) return;
  int ns = npos[src[e]];
  int nd = npos[dst[e]];
  float m = FIRST ? 1.0f : mask[e];
  m = (ns >= 0 && nd >= 0) ? m : 0.0f;
  nsrc[e]  = ns < 0 ? 0 : ns;
  ndst[e]  = nd < 0 ? 0 : nd;
  nmask[e] = m;
}

__global__ void readout(const h16* __restrict__ hft, float* __restrict__ xr, int k) {
  int b = blockIdx.x, f = threadIdx.x;     // block = 64 threads
  float mx = -3.402823466e+38f, sm = 0.0f;
  for (int i = 0; i < k; ++i) {
    float v = (float)hft[(b * k + i) * 64 + f];
    mx = fmaxf(mx, v);
    sm += v;
  }
  xr[b * 128 + f]      = mx;
  xr[b * 128 + 64 + f] = sm / (float)k;
}

// ---------------------------------------------------------------------------
// 64-wide edge aggregation: f16 gathers (half traffic), f32 atomic scatters
// ---------------------------------------------------------------------------
__global__ void sage_edge64(const h16* __restrict__ hft, const int* __restrict__ src,
                            const int* __restrict__ dst, const float* __restrict__ mask,
                            float* __restrict__ agg, float* __restrict__ cnt) {
  int e = blockIdx.x * blockDim.x + threadIdx.x;
  if (e >= E_) return;
  float m = mask[e];
  if (m == 0.0f) return;
  int s = src[e], d = dst[e];
  __builtin_prefetch(&hft[(size_t)s * 64], 0, 1);   // global_prefetch_b8
  atomicAdd(&cnt[d], m);
  #pragma unroll 8
  for (int f = 0; f < 64; ++f)
    atomicAdd(&agg[d * 64 + f], (float)hft[s * 64 + f] * m);
}

__global__ void mean_div(const float* __restrict__ agg, const float* __restrict__ cnt,
                         h16* __restrict__ mean, int total) {
  int i = blockIdx.x * blockDim.x + threadIdx.x;
  if (i >= total) return;
  mean[i] = (h16)(agg[i] / fmaxf(cnt[i >> 6], 1.0f));
}

__global__ void gcn_deg(const int* __restrict__ dst, const float* __restrict__ mask,
                        float* __restrict__ deg) {
  int e = blockIdx.x * blockDim.x + threadIdx.x;
  if (e >= E_) return;
  atomicAdd(&deg[dst[e]], mask[e]);
}

__global__ void dis_kernel(const float* __restrict__ deg, float* __restrict__ dis, int N) {
  int n = blockIdx.x * blockDim.x + threadIdx.x;
  if (n >= N) return;
  dis[n] = rsqrtf(1.0f + deg[n]);
}

__global__ void gcn_edge(const h16* __restrict__ xw, const int* __restrict__ src,
                         const int* __restrict__ dst, const float* __restrict__ mask,
                         const float* __restrict__ dis, float* __restrict__ acc) {
  int e = blockIdx.x * blockDim.x + threadIdx.x;
  if (e >= E_) return;
  float m = mask[e];
  if (m == 0.0f) return;
  int s = src[e], d = dst[e];
  __builtin_prefetch(&xw[(size_t)s * 64], 0, 1);
  float norm = dis[s] * m * dis[d];
  #pragma unroll 8
  for (int f = 0; f < 64; ++f)
    atomicAdd(&acc[d * 64 + f], (float)xw[s * 64 + f] * norm);
}

__global__ void gcn_final(const float* __restrict__ acc, const h16* __restrict__ xw,
                          const float* __restrict__ dis, const float* __restrict__ bias,
                          h16* __restrict__ out, int total) {
  int i = blockIdx.x * blockDim.x + threadIdx.x;
  if (i >= total) return;
  float di = dis[i >> 6];
  float v = acc[i] + (float)xw[i] * di * di + bias[i & 63];
  out[i] = (h16)fmaxf(v, 0.0f);
}

// ---------------------------------------------------------------------------
// WMMA GEMM: out[N,64] = act( A1 @ W1^T (+ A2 @ W2^T) (+ bias) ), all f16 in,
// f32 accumulate, f16 out. A/B fragments loaded straight as f16 (16-B chunks).
// block = 128 threads (4 waves); each wave owns one 16x16 output tile.
// Lane layout (16-bit 16x32 A): lane L -> row M=L&15, K group kg=(L>>4)*8,
// halves cover K={kg..kg+7} and {kg+16..kg+23}. B mirrors over columns.
// C: VGPR r -> M = r + 8*(L>>4), N = L&15. No divergence -> EXEC all-ones.
// ---------------------------------------------------------------------------
template <bool DUAL, bool RELU, bool BIAS>
__global__ __launch_bounds__(128)
void gemm64_wmma(const h16* __restrict__ A1, const h16* __restrict__ W1,
                 const h16* __restrict__ A2, const h16* __restrict__ W2,
                 const float* __restrict__ bias, h16* __restrict__ out) {
  const int lane = threadIdx.x & 31;
  const int wv   = threadIdx.x >> 5;            // 0..3 -> 16-col tile
  const int m    = lane & 15;
  const int kg   = (lane >> 4) << 3;            // 0 or 8
  const int row  = blockIdx.x * 16 + m;
  const int col  = wv * 16 + m;
  v8f c = {};
  {
    const h16* ap = A1 + (size_t)row * 64;
    const h16* bp = W1 + (size_t)col * 64;
    #pragma unroll
    for (int kk = 0; kk < 64; kk += 32) {
      v16h a, b;
      #pragma unroll
      for (int i = 0; i < 8; ++i) {
        a[i]     = ap[kk + kg + i];
        a[i + 8] = ap[kk + 16 + kg + i];
        b[i]     = bp[kk + kg + i];
        b[i + 8] = bp[kk + 16 + kg + i];
      }
      c = __builtin_amdgcn_wmma_f32_16x16x32_f16(false, a, false, b, (short)0, c,
                                                 false, false);
    }
  }
  if (DUAL) {
    const h16* ap = A2 + (size_t)row * 64;
    const h16* bp = W2 + (size_t)col * 64;
    #pragma unroll
    for (int kk = 0; kk < 64; kk += 32) {
      v16h a, b;
      #pragma unroll
      for (int i = 0; i < 8; ++i) {
        a[i]     = ap[kk + kg + i];
        a[i + 8] = ap[kk + 16 + kg + i];
        b[i]     = bp[kk + kg + i];
        b[i + 8] = bp[kk + 16 + kg + i];
      }
      c = __builtin_amdgcn_wmma_f32_16x16x32_f16(false, a, false, b, (short)0, c,
                                                 false, false);
    }
  }
  const int ocol  = wv * 16 + (lane & 15);
  const int rbase = blockIdx.x * 16 + ((lane >> 4) << 3);
  #pragma unroll
  for (int r = 0; r < 8; ++r) {
    float v = c[r];
    if (BIAS) v += bias[ocol];
    if (RELU) v = fmaxf(v, 0.0f);
    out[(size_t)(rbase + r) * 64 + ocol] = (h16)v;
  }
}

// ---------------------------------------------------------------------------
// Head: z = x1+x2+x4+x5 ; 3-layer MLP ; softmax. One block.
// ---------------------------------------------------------------------------
__global__ void mlp_head(const float* __restrict__ x1, const float* __restrict__ x2,
                         const float* __restrict__ x4, const float* __restrict__ x5,
                         const float* __restrict__ Wa, const float* __restrict__ ba,
                         const float* __restrict__ Wb, const float* __restrict__ bb,
                         const float* __restrict__ Wc, const float* __restrict__ bc,
                         float* __restrict__ out) {
  __shared__ float z[8 * 128], a1[8 * 128], a2[8 * 64], lg[8 * 256];
  __shared__ float rmax[8], rsum[8];
  int t = threadIdx.x;
  for (int i = t; i < 8 * 128; i += 256) z[i] = x1[i] + x2[i] + x4[i] + x5[i];
  __syncthreads();
  for (int i = t; i < 8 * 128; i += 256) {
    int b = i >> 7, j = i & 127;
    float v = ba[j];
    for (int k = 0; k < 128; ++k) v += z[b * 128 + k] * Wa[j * 128 + k];
    a1[i] = fmaxf(v, 0.0f);
  }
  __syncthreads();
  for (int i = t; i < 8 * 64; i += 256) {
    int b = i >> 6, j = i & 63;
    float v = bb[j];
    for (int k = 0; k < 128; ++k) v += a1[b * 128 + k] * Wb[j * 128 + k];
    a2[i] = fmaxf(v, 0.0f);
  }
  __syncthreads();
  for (int i = t; i < 8 * 256; i += 256) {
    int b = i >> 8, j = i & 255;
    float v = bc[j];
    for (int k = 0; k < 64; ++k) v += a2[b * 64 + k] * Wc[j * 64 + k];
    lg[i] = v;
  }
  __syncthreads();
  if (t < 8) {
    float mx = -3.402823466e+38f;
    for (int j = 0; j < 256; ++j) mx = fmaxf(mx, lg[t * 256 + j]);
    rmax[t] = mx;
  }
  __syncthreads();
  for (int i = t; i < 8 * 256; i += 256) lg[i] = __expf(lg[i] - rmax[i >> 8]);
  __syncthreads();
  if (t < 8) {
    float sm = 0.0f;
    for (int j = 0; j < 256; ++j) sm += lg[t * 256 + j];
    rsum[t] = sm;
  }
  __syncthreads();
  for (int i = t; i < 8 * 256; i += 256) out[i] = lg[i] / rsum[i >> 8];
}

// ---------------------------------------------------------------------------
extern "C" void kernel_launch(void* const* d_in, const int* in_sizes, int n_in,
                              void* d_out, int out_size, void* d_ws, size_t ws_size,
                              hipStream_t stream) {
  const float* x   = (const float*)d_in[0];
  const int*   ei  = (const int*)d_in[1];
  const float* Wl1 = (const float*)d_in[2];
  const float* bl1 = (const float*)d_in[3];
  const float* Wr1 = (const float*)d_in[4];
  const float* Wl2 = (const float*)d_in[5];
  const float* bl2 = (const float*)d_in[6];
  const float* Wr2 = (const float*)d_in[7];
  const float* W4  = (const float*)d_in[8];
  const float* b4  = (const float*)d_in[9];
  const float* W5  = (const float*)d_in[10];
  const float* b5  = (const float*)d_in[11];
  const float* wp1 = (const float*)d_in[12];
  const float* wp2 = (const float*)d_in[13];
  const float* wp4 = (const float*)d_in[14];
  const float* wp5 = (const float*)d_in[15];
  const float* Wa  = (const float*)d_in[16];
  const float* ba  = (const float*)d_in[17];
  const float* Wb  = (const float*)d_in[18];
  const float* bb  = (const float*)d_in[19];
  const float* Wc  = (const float*)d_in[20];
  const float* bc  = (const float*)d_in[21];
  const int* src0 = ei;
  const int* dst0 = ei + E_;

  // ---- workspace carve (256B aligned) ----
  char* p = (char*)d_ws;
  auto carve = [&](size_t bytes) -> void* {
    void* r = (void*)p;
    p += (bytes + 255) & ~(size_t)255;
    return r;
  };
  h16*   hA16 = (h16*)carve((size_t)N0_ * 64 * 2);
  h16*   hB16 = (h16*)carve((size_t)N0_ * 64 * 2);
  h16*   hC16 = (h16*)carve((size_t)N0_ * 64 * 2);
  float* fscr = (float*)carve((size_t)N1_ * 64 * 4);   // f32 atomic accumulator
  float* cnt  = (float*)carve((size_t)N0_ * 4);
  float* dis  = (float*)carve((size_t)N0_ * 4);
  float* scr  = (float*)carve((size_t)N0_ * 4);
  u64*   keys = (u64*)carve((size_t)N0_ * 8);
  int*   rnk  = (int*)carve((size_t)N0_ * 4);
  int*   nps  = (int*)carve((size_t)N0_ * 4);
  int*   srcB = (int*)carve((size_t)E_ * 4);
  int*   dstB = (int*)carve((size_t)E_ * 4);
  float* mskA = (float*)carve((size_t)E_ * 4);
  int*   srcC = (int*)carve((size_t)E_ * 4);
  int*   dstC = (int*)carve((size_t)E_ * 4);
  float* mskB = (float*)carve((size_t)E_ * 4);
  h16*   wl2h = (h16*)carve(64 * 64 * 2);
  h16*   wr2h = (h16*)carve(64 * 64 * 2);
  h16*   w4h  = (h16*)carve(64 * 64 * 2);
  h16*   w5h  = (h16*)carve(64 * 64 * 2);
  float* x1b  = (float*)carve(B_ * 128 * 4);
  float* x2b  = (float*)carve(B_ * 128 * 4);
  float* x4b  = (float*)carve(B_ * 128 * 4);
  float* x5b  = (float*)carve(B_ * 128 * 4);

  const int EB = E_ / 256;

  // ---- weights -> f16 (once) ----
  cvt_w64<<<16, 256, 0, stream>>>(Wl2, wl2h);
  cvt_w64<<<16, 256, 0, stream>>>(Wr2, wr2h);
  cvt_w64<<<16, 256, 0, stream>>>(W4, w4h);
  cvt_w64<<<16, 256, 0, stream>>>(W5, w5h);

  // ---- SAGE1 (N0, in-dim 2) -> hA16 ----
  hipMemsetAsync(fscr, 0, (size_t)N0_ * 2 * 4, stream);
  hipMemsetAsync(cnt, 0, (size_t)N0_ * 4, stream);
  sage1_edge<<<EB, 256, 0, stream>>>(x, src0, dst0, fscr, cnt);
  sage1_node<<<N0_ / 256, 256, 0, stream>>>(x, fscr, cnt, Wl1, bl1, Wr1, hA16);

  // ---- TopK1 (16384->8192) hA->hB, edges: input -> B set ----
  score_kernel<<<dim3(NG_ / 256, B_), 256, 0, stream>>>(hA16, wp1, scr, keys, NG_);
  rank_kernel<<<dim3(NG_ / 256, B_), 256, 0, stream>>>(keys, rnk, NG_);
  topk_compact<<<dim3(NG_ / 256, B_), 256, 0, stream>>>(hA16, scr, rnk, hB16, nps, NG_,
                                                        NG_ / 2);
  edge_remap<true><<<EB, 256, 0, stream>>>(src0, dst0, nullptr, nps, srcB, dstB, mskA);
  readout<<<B_, 64, 0, stream>>>(hB16, x1b, NG_ / 2);

  // ---- SAGE2 (N1) hB->hA, WMMA dual GEMM ----
  hipMemsetAsync(fscr, 0, (size_t)N1_ * 64 * 4, stream);
  hipMemsetAsync(cnt, 0, (size_t)N1_ * 4, stream);
  sage_edge64<<<EB, 256, 0, stream>>>(hB16, srcB, dstB, mskA, fscr, cnt);
  mean_div<<<(N1_ * 64) / 256, 256, 0, stream>>>(fscr, cnt, hC16, N1_ * 64);
  gemm64_wmma<true, true, true><<<N1_ / 16, 128, 0, stream>>>(hC16, wl2h, hB16, wr2h,
                                                              bl2, hA16);

  // ---- TopK2 (8192->4096) hA->hB, edges B -> C ----
  score_kernel<<<dim3(8192 / 256, B_), 256, 0, stream>>>(hA16, wp2, scr, keys, 8192);
  rank_kernel<<<dim3(8192 / 256, B_), 256, 0, stream>>>(keys, rnk, 8192);
  topk_compact<<<dim3(8192 / 256, B_), 256, 0, stream>>>(hA16, scr, rnk, hB16, nps, 8192,
                                                         4096);
  edge_remap<false><<<EB, 256, 0, stream>>>(srcB, dstB, mskA, nps, srcC, dstC, mskB);
  readout<<<B_, 64, 0, stream>>>(hB16, x2b, 4096);

  // ---- GCN4 (N2) input hB, edges C; xw in hC16, acc in fscr, out hA16 ----
  gemm64_wmma<false, false, false><<<N2_ / 16, 128, 0, stream>>>(hB16, w4h, nullptr,
                                                                 nullptr, nullptr, hC16);
  hipMemsetAsync(cnt, 0, (size_t)N2_ * 4, stream);
  gcn_deg<<<EB, 256, 0, stream>>>(dstC, mskB, cnt);
  dis_kernel<<<N2_ / 256, 256, 0, stream>>>(cnt, dis, N2_);
  hipMemsetAsync(fscr, 0, (size_t)N2_ * 64 * 4, stream);
  gcn_edge<<<EB, 256, 0, stream>>>(hC16, srcC, dstC, mskB, dis, fscr);
  gcn_final<<<(N2_ * 64) / 256, 256, 0, stream>>>(fscr, hC16, dis, b4, hA16, N2_ * 64);

  // ---- TopK4 (4096->2048) hA->hB, edges C -> B ----
  score_kernel<<<dim3(4096 / 256, B_), 256, 0, stream>>>(hA16, wp4, scr, keys, 4096);
  rank_kernel<<<dim3(4096 / 256, B_), 256, 0, stream>>>(keys, rnk, 4096);
  topk_compact<<<dim3(4096 / 256, B_), 256, 0, stream>>>(hA16, scr, rnk, hB16, nps, 4096,
                                                         2048);
  edge_remap<false><<<EB, 256, 0, stream>>>(srcC, dstC, mskB, nps, srcB, dstB, mskA);
  readout<<<B_, 64, 0, stream>>>(hB16, x4b, 2048);

  // ---- GCN5 (N4) input hB, edges B ----
  gemm64_wmma<false, false, false><<<N4_ / 16, 128, 0, stream>>>(hB16, w5h, nullptr,
                                                                 nullptr, nullptr, hC16);
  hipMemsetAsync(cnt, 0, (size_t)N4_ * 4, stream);
  gcn_deg<<<EB, 256, 0, stream>>>(dstB, mskA, cnt);
  dis_kernel<<<N4_ / 256, 256, 0, stream>>>(cnt, dis, N4_);
  hipMemsetAsync(fscr, 0, (size_t)N4_ * 64 * 4, stream);
  gcn_edge<<<EB, 256, 0, stream>>>(hC16, srcB, dstB, mskA, dis, fscr);
  gcn_final<<<(N4_ * 64) / 256, 256, 0, stream>>>(fscr, hC16, dis, b5, hA16, N4_ * 64);

  // ---- TopK5 (2048->1024) hA->hB (no edge remap needed after last pool) ----
  score_kernel<<<dim3(2048 / 256, B_), 256, 0, stream>>>(hA16, wp5, scr, keys, 2048);
  rank_kernel<<<dim3(2048 / 256, B_), 256, 0, stream>>>(keys, rnk, 2048);
  topk_compact<<<dim3(2048 / 256, B_), 256, 0, stream>>>(hA16, scr, rnk, hB16, nps, 2048,
                                                         1024);
  readout<<<B_, 64, 0, stream>>>(hB16, x5b, 1024);

  // ---- head ----
  mlp_head<<<1, 256, 0, stream>>>(x1b, x2b, x4b, x5b, Wa, ba, Wb, bb, Wc, bc,
                                  (float*)d_out);
}
